// GAEncoder_Decoder_20529943674886
// MI455X (gfx1250) — matrix operand order repaired and checked by reading, not compiled, not run on hardware
//
#include <hip/hip_runtime.h>
#include <hip/hip_bf16.h>
#include <stdint.h>

typedef __attribute__((ext_vector_type(16))) __bf16 v16bf;
typedef __attribute__((ext_vector_type(8)))  float  v8f;

#define N_NODES 20000
#define IN_DIM  2000
#define HID     256
#define OUTD    64

#if __has_builtin(__builtin_amdgcn_global_load_async_to_lds_b128) && __has_builtin(__builtin_amdgcn_s_wait_asynccnt)
#define USE_ASYNC_LDS 1
#else
#define USE_ASYNC_LDS 0
#endif

__device__ __forceinline__ unsigned short f2bf(float f) {
  unsigned int u = __builtin_bit_cast(unsigned int, f);
  u += 0x7FFFu + ((u >> 16) & 1u);   // round-to-nearest-even
  return (unsigned short)(u >> 16);
}

// copy 16 bytes global -> LDS (async on CDNA5, sync fallback otherwise)
typedef int v4i32 __attribute__((vector_size(4 * sizeof(int))));
typedef __attribute__((address_space(1))) v4i32* gbl_v4i_ptr;
typedef __attribute__((address_space(3))) v4i32* lds_v4i_ptr;

__device__ __forceinline__ void cp16_g2l(const unsigned short* g, unsigned short* l) {
#if USE_ASYNC_LDS
  __builtin_amdgcn_global_load_async_to_lds_b128(
      (gbl_v4i_ptr)(void*)(g),
      (lds_v4i_ptr)(void*)(l), 0, 0);
#else
  *(uint4*)l = *(const uint4*)g;
#endif
}

union Frag { unsigned int u[8]; v16bf v; };

// ------------------------- precision conversion -------------------------
__global__ void cast_bf16_kernel(const float* __restrict__ in, unsigned short* __restrict__ out, int n4) {
  int i = blockIdx.x * blockDim.x + threadIdx.x;
  if (i >= n4) return;
  const float4 v = ((const float4*)in)[i];
  ushort4 o;
  o.x = f2bf(v.x); o.y = f2bf(v.y); o.z = f2bf(v.z); o.w = f2bf(v.w);
  ((ushort4*)out)[i] = o;
}

// in: fp32 [K][N]  ->  out: bf16 [N][K]
__global__ void transpose_cast_kernel(const float* __restrict__ in, unsigned short* __restrict__ out,
                                      int K, int N) {
  int idx = blockIdx.x * blockDim.x + threadIdx.x;
  if (idx >= K * N) return;
  int n = idx / K;
  int k = idx - n * K;
  out[idx] = f2bf(in[(size_t)k * N + n]);
}

// ------------------------- graph normalization -------------------------
__global__ void deg_init_kernel(float* deg, int n) {
  int i = blockIdx.x * blockDim.x + threadIdx.x;
  if (i < n) deg[i] = 1.0f;                    // self-loop contribution
}

__global__ void deg_count_kernel(const int* __restrict__ dst, float* deg, int e) {
  int i = blockIdx.x * blockDim.x + threadIdx.x;
  if (i < e) atomicAdd(&deg[dst[i]], 1.0f);
}

__global__ void dinv_kernel(const float* __restrict__ deg, float* dinv, int n) {
  int i = blockIdx.x * blockDim.x + threadIdx.x;
  if (i < n) {
    float d = deg[i];
    dinv[i] = d > 0.0f ? rsqrtf(d) : 0.0f;
  }
}

__global__ void norm_kernel(const int* __restrict__ src, const int* __restrict__ dst,
                            const float* __restrict__ dinv, float* norm, int e) {
  int i = blockIdx.x * blockDim.x + threadIdx.x;
  if (i < e) norm[i] = dinv[src[i]] * dinv[dst[i]];
}

// --------------------- scatter-aggregate (segment_sum) ---------------------
__global__ void agg_init_kernel(const float* __restrict__ xw, const float* __restrict__ dinv,
                                float* __restrict__ agg, int total, int cshift) {
  int idx = blockIdx.x * blockDim.x + threadIdx.x;
  if (idx >= total) return;
  int i = idx >> cshift;
  float w = dinv[i];
  agg[idx] = w * w * xw[idx];
}

__global__ void agg_scatter_kernel(const int* __restrict__ src, const int* __restrict__ dst,
                                   const float* __restrict__ norm, const float* __restrict__ xw,
                                   float* __restrict__ agg, int e, int gshift, int C) {
  int idx = blockIdx.x * blockDim.x + threadIdx.x;
  int eid = idx >> gshift;
  if (eid >= e) return;
  int g = idx & ((1 << gshift) - 1);
  int s = src[eid], d = dst[eid];
  float w = norm[eid];
  const float4 x = *((const float4*)(xw + (size_t)s * C) + g);
  float* o = agg + (size_t)d * C + (g << 2);
  atomicAdd(o + 0, w * x.x);
  atomicAdd(o + 1, w * x.y);
  atomicAdd(o + 2, w * x.z);
  atomicAdd(o + 3, w * x.w);
}

// v = act(agg + bc + xs + b); optionally store fp32 and/or bf16
__global__ void combine_kernel(const float* __restrict__ agg, const float* __restrict__ xs,
                               const float* __restrict__ bc, const float* __restrict__ b,
                               float* __restrict__ outf, unsigned short* __restrict__ outb,
                               int total, int cmask, int relu) {
  int idx = blockIdx.x * blockDim.x + threadIdx.x;
  if (idx >= total) return;
  int c = idx & cmask;
  float v = agg[idx] + bc[c] + xs[idx] + b[c];
  if (relu) v = fmaxf(v, 0.0f);
  if (outf) outf[idx] = v;
  if (outb) outb[idx] = f2bf(v);
}

// wave-per-row log_softmax over 64 columns (wave32: 2 cols/lane); fp32 in, bf16 out
__global__ __launch_bounds__(256) void log_softmax64_kernel(const float* __restrict__ z0,
                                                            unsigned short* __restrict__ zb, int nrows) {
  int wave = threadIdx.x >> 5;
  int lane = threadIdx.x & 31;
  int row = blockIdx.x * 8 + wave;
  if (row >= nrows) return;
  const float* p = z0 + (size_t)row * 64;
  float a = p[lane], b = p[lane + 32];
  float m = fmaxf(a, b);
  for (int off = 16; off; off >>= 1) m = fmaxf(m, __shfl_xor(m, off, 32));
  float s = __expf(a - m) + __expf(b - m);
  for (int off = 16; off; off >>= 1) s += __shfl_xor(s, off, 32);
  float ls = __logf(s) + m;
  unsigned short* q = zb + (size_t)row * 64;
  q[lane]      = f2bf(a - ls);
  q[lane + 32] = f2bf(b - ls);
}

// ------------------------- bf16 WMMA GEMM -------------------------
// C[M,N] = act(A[M,K] @ B[K,N] + bias)
// A: bf16 [M,K] row-major.  Bt: bf16 [N,K] (pre-transposed).  C: fp32 or bf16.
// Block tile 64x128, 256 threads = 8 waves (2x4), wave tile 32x32 (2x2 WMMA).
// Double-buffered LDS, async global->LDS copies on full interior tiles.
#define APITCH 40   // shorts per row (64B data + pad) -> 16B aligned chunks, conflict-free ds reads

__global__ __launch_bounds__(256) void wmma_gemm_kernel(
    const unsigned short* __restrict__ A, const unsigned short* __restrict__ Bt,
    const float* __restrict__ bias,
    float* __restrict__ Cf, unsigned short* __restrict__ Cb,
    int M, int N, int K, int relu)
{
  __shared__ __align__(16) unsigned short As[2][64 * APITCH];
  __shared__ __align__(16) unsigned short Bs[2][128 * APITCH];

  const int tid   = threadIdx.x;
  const int lane  = tid & 31;
  const int wave  = tid >> 5;
  const int waveM = wave >> 2;              // 0..1
  const int waveN = wave & 3;               // 0..3
  const int half  = lane >> 4;
  const int l16   = lane & 15;

  const int blockRow = blockIdx.y * 64;
  const int blockCol = blockIdx.x * 128;
  const bool blkFull = (blockRow + 64 <= M) && (blockCol + 128 <= N);

  // staging coords: A = 64 rows x 4 16B-chunks (1/thread); B = 128 rows x 4 (2/thread)
  const int ar = tid >> 2, ac = tid & 3;

  v8f acc[2][2];
#pragma unroll
  for (int i = 0; i < 2; ++i)
#pragma unroll
    for (int j = 0; j < 2; ++j) acc[i][j] = 0.0f;

  const int nk = (K + 31) >> 5;

  // full-tile stage: 3 async copies per thread, unconditional
  auto stage_full = [&](int it, int buf) {
    const int k0 = it << 5;
    cp16_g2l(A + (size_t)(blockRow + ar) * K + k0 + ac * 8,
             &As[buf][ar * APITCH + ac * 8]);
#pragma unroll
    for (int t = 0; t < 2; ++t) {
      int idx = tid + t * 256;
      int br = idx >> 2, bc2 = idx & 3;
      cp16_g2l(Bt + (size_t)(blockCol + br) * K + k0 + bc2 * 8,
               &Bs[buf][br * APITCH + bc2 * 8]);
    }
  };

  // edge-tile stage: guarded scalar fill (no asyncs issued)
  auto stage_partial = [&](int it, int buf) {
    const int k0 = it << 5;
#pragma unroll
    for (int t = 0; t < 8; ++t) {
      int idx = tid + t * 256;            // 64*32 shorts
      int r = idx >> 5, kk = idx & 31;
      int gr = blockRow + r, gk = k0 + kk;
      As[buf][r * APITCH + kk] = (gr < M && gk < K) ? A[(size_t)gr * K + gk] : (unsigned short)0;
    }
#pragma unroll
    for (int t = 0; t < 16; ++t) {
      int idx = tid + t * 256;            // 128*32 shorts
      int r = idx >> 5, kk = idx & 31;
      int gn = blockCol + r, gk = k0 + kk;
      Bs[buf][r * APITCH + kk] = (gn < N && gk < K) ? Bt[(size_t)gn * K + gk] : (unsigned short)0;
    }
  };

  // prologue: stage tile 0
  if (blkFull && 32 <= K) stage_full(0, 0); else stage_partial(0, 0);

  for (int it = 0; it < nk; ++it) {
    const int cur = it & 1;
    bool nextAsync = false;
    if (it + 1 < nk) {
      const bool f = blkFull && (((it + 1) << 5) + 32 <= K);
      if (f) { stage_full(it + 1, cur ^ 1); nextAsync = true; }
      else   { stage_partial(it + 1, cur ^ 1); }
    }
#if USE_ASYNC_LDS
    if (nextAsync) __builtin_amdgcn_s_wait_asynccnt(3);   // keep next tile's 3 in flight
    else           __builtin_amdgcn_s_wait_asynccnt(0);
#else
    (void)nextAsync;
#endif
    __syncthreads();

    // A fragment (16-bit 16x32): lane m=l16; VGPR v: K = 2*(v&3) + 8*half + 16*(v>>2)
    Frag fa[2];
#pragma unroll
    for (int tm = 0; tm < 2; ++tm) {
      const unsigned short* ap = &As[cur][(waveM * 32 + tm * 16 + l16) * APITCH];
#pragma unroll
      for (int v = 0; v < 8; ++v) {
        int kb = ((v & 3) << 1) + (half << 3) + ((v >> 2) << 4);
        fa[tm].u[v] = *(const unsigned int*)(ap + kb);
      }
    }
    // B fragment (32x16): lane n=l16; VGPR v: K = 2*(v&3) + 8*half + 16*(v>>2)
    Frag fb[2];
#pragma unroll
    for (int tn = 0; tn < 2; ++tn) {
      const unsigned short* bp = &Bs[cur][(waveN * 32 + tn * 16 + l16) * APITCH];
#pragma unroll
      for (int v = 0; v < 8; ++v) {
        int kb = ((v & 3) << 1) + (half << 3) + ((v >> 2) << 4);
        fb[tn].u[v] = *(const unsigned int*)(bp + kb);
      }
    }

#pragma unroll
    for (int tm = 0; tm < 2; ++tm)
#pragma unroll
      for (int tn = 0; tn < 2; ++tn)
        acc[tm][tn] = __builtin_amdgcn_wmma_f32_16x16x32_bf16(
            false, fa[tm].v, false, fb[tn].v, (short)0, acc[tm][tn], false, false);

    __syncthreads();
  }

  // epilogue: row = base + r + 8*half, col = base + l16
#pragma unroll
  for (int tm = 0; tm < 2; ++tm) {
#pragma unroll
    for (int tn = 0; tn < 2; ++tn) {
      int col = blockCol + waveN * 32 + tn * 16 + l16;
      if (col < N) {
        float bv = bias ? bias[col] : 0.0f;
        int rowBase = blockRow + waveM * 32 + tm * 16 + (half << 3);
#pragma unroll
        for (int r = 0; r < 8; ++r) {
          int row = rowBase + r;
          if (row < M) {
            float v = acc[tm][tn][r] + bv;
            if (relu) v = fmaxf(v, 0.0f);
            if (Cb) Cb[(size_t)row * N + col] = f2bf(v);
            else    Cf[(size_t)row * N + col] = v;
          }
        }
      }
    }
  }
}

// ------------------------- host orchestration -------------------------
static inline void launch_gemm(const unsigned short* A, const unsigned short* Bt, const float* bias,
                               float* Cf, unsigned short* Cb, int M, int N, int K, int relu,
                               hipStream_t s) {
  dim3 grid((N + 127) / 128, (M + 63) / 64);
  wmma_gemm_kernel<<<grid, 256, 0, s>>>(A, Bt, bias, Cf, Cb, M, N, K, relu);
}

extern "C" void kernel_launch(void* const* d_in, const int* in_sizes, int n_in,
                              void* d_out, int out_size, void* d_ws, size_t ws_size,
                              hipStream_t stream) {
  (void)n_in; (void)out_size; (void)ws_size;
  const float* x   = (const float*)d_in[0];
  const int*   ei  = (const int*)d_in[1];
  const float* Wc1 = (const float*)d_in[2];  const float* bc1 = (const float*)d_in[3];
  const float* Wc2 = (const float*)d_in[4];  const float* bc2 = (const float*)d_in[5];
  const float* Wc3 = (const float*)d_in[6];  const float* bc3 = (const float*)d_in[7];
  const float* W1  = (const float*)d_in[8];  const float* b1  = (const float*)d_in[9];
  const float* W2  = (const float*)d_in[10]; const float* b2  = (const float*)d_in[11];
  const float* W3  = (const float*)d_in[12]; const float* b3  = (const float*)d_in[13];
  const float* W4  = (const float*)d_in[14]; const float* b4  = (const float*)d_in[15];
  const float* W5  = (const float*)d_in[16]; const float* b5  = (const float*)d_in[17];
  const float* W6  = (const float*)d_in[18]; const float* b6  = (const float*)d_in[19];
  float* out = (float*)d_out;

  const int E = in_sizes[1] / 2;
  const int* src = ei;
  const int* dst = ei + E;

  // workspace carve (256B aligned)
  char* base = (char*)d_ws;
  size_t off = 0;
  auto carve = [&](size_t bytes) { void* p = base + off; off += (bytes + 255) & ~(size_t)255; return p; };
  float* deg  = (float*)carve(sizeof(float) * N_NODES);
  float* dinv = (float*)carve(sizeof(float) * N_NODES);
  float* nrm  = (float*)carve(sizeof(float) * E);
  float* xw   = (float*)carve(sizeof(float) * (size_t)N_NODES * HID);
  float* xs   = (float*)carve(sizeof(float) * (size_t)N_NODES * HID);
  float* agg  = (float*)carve(sizeof(float) * (size_t)N_NODES * HID);
  float* z0   = (float*)carve(sizeof(float) * (size_t)N_NODES * OUTD);
  unsigned short* xb   = (unsigned short*)carve(2ull * N_NODES * IN_DIM);
  unsigned short* h1b  = (unsigned short*)carve(2ull * N_NODES * HID);
  unsigned short* h2b  = (unsigned short*)carve(2ull * N_NODES * HID);
  unsigned short* zb   = (unsigned short*)carve(2ull * N_NODES * OUTD);
  unsigned short* Wc1t = (unsigned short*)carve(2ull * IN_DIM * HID);
  unsigned short* W1t  = (unsigned short*)carve(2ull * IN_DIM * HID);
  unsigned short* Wc2t = (unsigned short*)carve(2ull * HID * HID);
  unsigned short* W2t  = (unsigned short*)carve(2ull * HID * HID);
  unsigned short* Wc3t = (unsigned short*)carve(2ull * HID * OUTD);
  unsigned short* W3t  = (unsigned short*)carve(2ull * HID * OUTD);
  unsigned short* W4t  = (unsigned short*)carve(2ull * OUTD * HID);
  unsigned short* W5t  = (unsigned short*)carve(2ull * HID * HID);
  unsigned short* W6t  = (unsigned short*)carve(2ull * HID * IN_DIM);
  unsigned short* d1b = h1b;   // h1 dead after layer-2 GEMMs
  unsigned short* d2b = h2b;   // h2 dead after layer-3 GEMMs

  const int TPB = 256;
  auto blocks = [](long long n) { return (unsigned)((n + 255) / 256); };

  // ---- one-time precision prep ----
  cast_bf16_kernel<<<blocks((long long)N_NODES * IN_DIM / 4), TPB, 0, stream>>>(x, xb, N_NODES * IN_DIM / 4);
  transpose_cast_kernel<<<blocks((long long)IN_DIM * HID), TPB, 0, stream>>>(Wc1, Wc1t, IN_DIM, HID);
  transpose_cast_kernel<<<blocks((long long)IN_DIM * HID), TPB, 0, stream>>>(W1,  W1t,  IN_DIM, HID);
  transpose_cast_kernel<<<blocks((long long)HID * HID), TPB, 0, stream>>>(Wc2, Wc2t, HID, HID);
  transpose_cast_kernel<<<blocks((long long)HID * HID), TPB, 0, stream>>>(W2,  W2t,  HID, HID);
  transpose_cast_kernel<<<blocks((long long)HID * OUTD), TPB, 0, stream>>>(Wc3, Wc3t, HID, OUTD);
  transpose_cast_kernel<<<blocks((long long)HID * OUTD), TPB, 0, stream>>>(W3,  W3t,  HID, OUTD);
  transpose_cast_kernel<<<blocks((long long)OUTD * HID), TPB, 0, stream>>>(W4,  W4t,  OUTD, HID);
  transpose_cast_kernel<<<blocks((long long)HID * HID), TPB, 0, stream>>>(W5,  W5t,  HID, HID);
  transpose_cast_kernel<<<blocks((long long)HID * IN_DIM), TPB, 0, stream>>>(W6,  W6t,  HID, IN_DIM);

  // ---- GCN symmetric normalization ----
  deg_init_kernel<<<blocks(N_NODES), TPB, 0, stream>>>(deg, N_NODES);
  deg_count_kernel<<<blocks(E), TPB, 0, stream>>>(dst, deg, E);
  dinv_kernel<<<blocks(N_NODES), TPB, 0, stream>>>(deg, dinv, N_NODES);
  norm_kernel<<<blocks(E), TPB, 0, stream>>>(src, dst, dinv, nrm, E);

  // ---- layer 1: h1 = relu(gcn(x,Wc1)+bc1 + x@W1+b1) ----
  launch_gemm(xb, Wc1t, nullptr, xw, nullptr, N_NODES, HID, IN_DIM, 0, stream);
  launch_gemm(xb, W1t,  nullptr, xs, nullptr, N_NODES, HID, IN_DIM, 0, stream);
  agg_init_kernel<<<blocks((long long)N_NODES * HID), TPB, 0, stream>>>(xw, dinv, agg, N_NODES * HID, 8);
  agg_scatter_kernel<<<blocks((long long)E * (HID / 4)), TPB, 0, stream>>>(src, dst, nrm, xw, agg, E, 6, HID);
  combine_kernel<<<blocks((long long)N_NODES * HID), TPB, 0, stream>>>(agg, xs, bc1, b1, nullptr, h1b, N_NODES * HID, HID - 1, 1);

  // ---- layer 2 ----
  launch_gemm(h1b, Wc2t, nullptr, xw, nullptr, N_NODES, HID, HID, 0, stream);
  launch_gemm(h1b, W2t,  nullptr, xs, nullptr, N_NODES, HID, HID, 0, stream);
  agg_init_kernel<<<blocks((long long)N_NODES * HID), TPB, 0, stream>>>(xw, dinv, agg, N_NODES * HID, 8);
  agg_scatter_kernel<<<blocks((long long)E * (HID / 4)), TPB, 0, stream>>>(src, dst, nrm, xw, agg, E, 6, HID);
  combine_kernel<<<blocks((long long)N_NODES * HID), TPB, 0, stream>>>(agg, xs, bc2, b2, nullptr, h2b, N_NODES * HID, HID - 1, 1);

  // ---- layer 3 + log_softmax ----
  launch_gemm(h2b, Wc3t, nullptr, xw, nullptr, N_NODES, OUTD, HID, 0, stream);
  launch_gemm(h2b, W3t,  nullptr, xs, nullptr, N_NODES, OUTD, HID, 0, stream);
  agg_init_kernel<<<blocks((long long)N_NODES * OUTD), TPB, 0, stream>>>(xw, dinv, agg, N_NODES * OUTD, 6);
  agg_scatter_kernel<<<blocks((long long)E * (OUTD / 4)), TPB, 0, stream>>>(src, dst, nrm, xw, agg, E, 4, OUTD);
  combine_kernel<<<blocks((long long)N_NODES * OUTD), TPB, 0, stream>>>(agg, xs, bc3, b3, z0, nullptr, N_NODES * OUTD, OUTD - 1, 0);
  log_softmax64_kernel<<<(N_NODES + 7) / 8, TPB, 0, stream>>>(z0, zb, N_NODES);

  // ---- decoder ----
  launch_gemm(zb,  W4t, b4, nullptr, d1b, N_NODES, HID, OUTD, 1, stream);
  launch_gemm(d1b, W5t, b5, nullptr, d2b, N_NODES, HID, HID, 1, stream);
  launch_gemm(d2b, W6t, b6, out, nullptr, N_NODES, IN_DIM, HID, 0, stream);
}
